// MaskedCrossAttention_56985626083552
// MI455X (gfx1250) — matrix operand order repaired and checked by reading, not compiled, not run on hardware
//
#include <hip/hip_runtime.h>
#include <stdint.h>

#define B_    4
#define T_    4096
#define L_    64
#define W_    8
#define DIM_  2048
#define DIMA_ 1024
#define H_    8
#define DH_   64
#define INNER_ 512

typedef __attribute__((ext_vector_type(16))) __bf16 v16bf;
typedef __attribute__((ext_vector_type(8)))  float  v8f;
typedef __attribute__((ext_vector_type(8)))  unsigned int v8u;

// ---------- helpers ----------
__device__ __forceinline__ unsigned short f2bf(float f) {
  unsigned int u = __builtin_bit_cast(unsigned int, f);
  u += 0x7FFFu + ((u >> 16) & 1u);          // round-to-nearest-even
  return (unsigned short)(u >> 16);
}

// CDNA5 async global->LDS copy (16 bytes), tracked by ASYNCcnt.
// GV mode: vdst = LDS byte offset (low 32 bits of generic LDS address),
//          vaddr = 64-bit global address, saddr = off.
__device__ __forceinline__ void async_ld16(void* lds_dst, const void* g) {
#if defined(__gfx1250__)
  unsigned off = (unsigned)(uintptr_t)lds_dst;
  asm volatile("global_load_async_to_lds_b128 %0, %1, off"
               :: "v"(off), "v"(g) : "memory");
#else
  *(uint4*)lds_dst = *(const uint4*)g;      // host-pass / other-arch parse only
#endif
}
__device__ __forceinline__ void wait_async0() {
#if defined(__gfx1250__)
  asm volatile("s_wait_asynccnt 0x0" ::: "memory");
#endif
}
__device__ __forceinline__ void wait_async2() {
#if defined(__gfx1250__)
  asm volatile("s_wait_asynccnt 0x2" ::: "memory");
#endif
}

// A-fragment (16x32 bf16, per ISA 7.12.2): lane m = lane&15, kg = lane>>4,
// VGPR 0-3 -> K = kg*8 + {0..7}, VGPR 4-7 -> K = 16 + kg*8 + {0..7} (packed pairs)
__device__ __forceinline__ v16bf fragA(const unsigned short* base, int rowStride,
                                       int mBase, int kOff) {
  int lane = threadIdx.x & 31;
  const unsigned int* p =
      (const unsigned int*)(base + (size_t)(mBase + (lane & 15)) * rowStride + kOff);
  int b0 = (lane >> 4) * 4;
  v8u u;
  u[0] = p[b0 + 0]; u[1] = p[b0 + 1]; u[2] = p[b0 + 2]; u[3] = p[b0 + 3];
  u[4] = p[b0 + 8]; u[5] = p[b0 + 9]; u[6] = p[b0 + 10]; u[7] = p[b0 + 11];
  return __builtin_bit_cast(v16bf, u);
}

// B-fragment (32x16 bf16): source stored transposed as [N][K] rows so the packed
// K pair (2v,2v+1) is one b32 read. lane n = lane&15, K half = lane>>4.
__device__ __forceinline__ v16bf fragB(const unsigned short* base, int rowStride,
                                       int nBase, int kOff) {
  int lane = threadIdx.x & 31;
  const unsigned int* p =
      (const unsigned int*)(base + (size_t)(nBase + (lane & 15)) * rowStride + kOff);
  int b0 = (lane >> 4) * 4;
  v8u u;
#pragma unroll
  for (int v = 0; v < 8; ++v) u[v] = p[b0 + (v & 3) + (v >> 2) * 8];
  return __builtin_bit_cast(v16bf, u);
}

__device__ __forceinline__ v8f wmma_bf16(v16bf a, v16bf b, v8f c) {
  return __builtin_amdgcn_wmma_f32_16x16x32_bf16(false, a, false, b, (short)0, c,
                                                 false, false);
}

// ---------- LayerNorm + bf16 pack ----------
__global__ __launch_bounds__(256) void ln_kernel(const float* __restrict__ x,
                                                 const float* __restrict__ gamma,
                                                 const float* __restrict__ beta,
                                                 unsigned short* __restrict__ xn) {
  int row = blockIdx.x;
  const float* xr = x + (size_t)row * DIM_;
  int tid = threadIdx.x;
  float v[8]; float s = 0.f, sq = 0.f;
#pragma unroll
  for (int i = 0; i < 8; ++i) { float t = xr[tid + i * 256]; v[i] = t; s += t; sq += t * t; }
#pragma unroll
  for (int off = 16; off >= 1; off >>= 1) {
    s += __shfl_xor(s, off, 32);
    sq += __shfl_xor(sq, off, 32);
  }
  __shared__ float red[16];
  int w = tid >> 5, lane = tid & 31;
  if (lane == 0) { red[w] = s; red[8 + w] = sq; }
  __syncthreads();
  if (tid == 0) {
    float a = 0.f, b = 0.f;
    for (int j = 0; j < 8; ++j) { a += red[j]; b += red[8 + j]; }
    red[0] = a; red[8] = b;
  }
  __syncthreads();
  float mean = red[0] * (1.0f / DIM_);
  float var  = red[8] * (1.0f / DIM_) - mean * mean;
  float rstd = rsqrtf(var + 1e-5f);
  unsigned short* orow = xn + (size_t)row * DIM_;
#pragma unroll
  for (int i = 0; i < 8; ++i) {
    int c = tid + i * 256;
    orow[c] = f2bf((v[i] - mean) * rstd * gamma[c] + beta[c]);
  }
}

// ---------- fp32 -> bf16 weight convert ----------
__global__ __launch_bounds__(256) void cvt_kernel(const float* __restrict__ in,
                                                  unsigned short* __restrict__ out, int n) {
  int i = blockIdx.x * 256 + threadIdx.x;
  if (i < n) out[i] = f2bf(in[i]);
}

// ---------- KV projection (small): media [B*L,1024] @ Wkv [1024,1024] ----------
__global__ __launch_bounds__(256) void kv_kernel(const float* __restrict__ media,
                                                 const float* __restrict__ Wkv,
                                                 unsigned short* __restrict__ kws,
                                                 unsigned short* __restrict__ vT) {
  int g = blockIdx.x * 256 + threadIdx.x;      // B*L*1024 outputs
  int c  = g & 1023;
  int bl = g >> 10;
  int b = bl >> 6, l = bl & 63;
  __shared__ float sm[DIMA_];
  const float* mrow = media + (size_t)bl * DIMA_;
  for (int i = threadIdx.x; i < DIMA_; i += 256) sm[i] = mrow[i];
  __syncthreads();
  float acc = 0.f;
#pragma unroll 4
  for (int i = 0; i < DIMA_; ++i) acc = fmaf(sm[i], Wkv[(size_t)i * 1024 + c], acc);
  unsigned short val = f2bf(acc);
  if (c < INNER_) {                            // K: [b][h][l][dh]
    int hh = c >> 6, dh = c & 63;
    kws[((((size_t)b * H_ + hh) * L_) + l) * DH_ + dh] = val;
  } else {                                     // V^T: [b][h][dh][l]
    int c2 = c - INNER_;
    int hh = c2 >> 6, dh = c2 & 63;
    vT[((((size_t)b * H_ + hh) * DH_) + dh) * L_ + l] = val;
  }
}

// ---------- window-end scan: end[b][t] = max(cumsum(loc),1)*W ----------
__global__ __launch_bounds__(256) void scan_kernel(const int* __restrict__ ml,
                                                   int* __restrict__ endi) {
  int b = blockIdx.x, tid = threadIdx.x;
  __shared__ int part[256];
  int loc[16]; int s = 0;
  int base = b * T_ + tid * 16;
#pragma unroll
  for (int i = 0; i < 16; ++i) { int m = (ml[base + i] != 0); loc[i] = m; s += m; }
  part[tid] = s;
  __syncthreads();
  for (int off = 1; off < 256; off <<= 1) {
    int v = (tid >= off) ? part[tid - off] : 0;
    __syncthreads();
    part[tid] += v;
    __syncthreads();
  }
  int run = (tid == 0) ? 0 : part[tid - 1];
#pragma unroll
  for (int i = 0; i < 16; ++i) {
    run += loc[i];
    endi[base + i] = (run < 1 ? 1 : run) * W_;
  }
}

// ---------- WMMA GEMM: C[M,N] = alpha * A[M,K](bf16) @ B[K,N](bf16) ----------
// block tile 128x64, BK=32, 256 threads = 8 waves (4x2), 2x2 WMMA tiles/wave.
// A tiles stream global->LDS via CDNA5 async-to-LDS, double buffered over K.
template <bool OUT_BF16>
__global__ __launch_bounds__(256) void gemm_kernel(const unsigned short* __restrict__ A,
                                                   const unsigned short* __restrict__ Bw,
                                                   void* __restrict__ Cp,
                                                   int M, int N, int K,
                                                   int lda, int ldb, int ldc, float alpha) {
  __shared__ __attribute__((aligned(16))) unsigned short sA[2][128 * 32];
  __shared__ __attribute__((aligned(16))) unsigned short sB[64 * 32];   // [n][k]
  int bn0 = blockIdx.x * 64, bm0 = blockIdx.y * 128;
  int tid = threadIdx.x;
  int w = tid >> 5, lane = tid & 31;
  int wm = w >> 1, wn = w & 1;

  v8f acc[4];
#pragma unroll
  for (int t = 0; t < 4; ++t)
#pragma unroll
    for (int r = 0; r < 8; ++r) acc[t][r] = 0.0f;

  int nk = K >> 5;

  // preload A tile 0: 512 x 16B chunks, 2 async ops per thread (per wave)
#pragma unroll
  for (int c = tid; c < 512; c += 256) {
    int row = c >> 2, kc = (c & 3) << 3;
    async_ld16(&sA[0][row * 32 + kc], A + (size_t)(bm0 + row) * lda + kc);
  }

  for (int kt = 0; kt < nk; ++kt) {
    int cur = kt & 1;
    // stage B (transposed into [n][k]) for this tile
#pragma unroll
    for (int e = tid; e < 2048; e += 256) {
      int n = e & 63, k = e >> 6;
      sB[n * 32 + k] = Bw[(size_t)(kt * 32 + k) * ldb + bn0 + n];
    }
    // issue async A for next tile into the other buffer
    if (kt + 1 < nk) {
#pragma unroll
      for (int c = tid; c < 512; c += 256) {
        int row = c >> 2, kc = (c & 3) << 3;
        async_ld16(&sA[cur ^ 1][row * 32 + kc],
                   A + (size_t)(bm0 + row) * lda + (kt + 1) * 32 + kc);
      }
      wait_async2();   // retire this tile's batch; keep next tile's in flight
    } else {
      wait_async0();
    }
    __syncthreads();

    v16bf a0 = fragA(sA[cur], 32, wm * 32, 0);
    v16bf a1 = fragA(sA[cur], 32, wm * 32 + 16, 0);
    v16bf b0 = fragB(sB, 32, wn * 32, 0);
    v16bf b1 = fragB(sB, 32, wn * 32 + 16, 0);
    acc[0] = wmma_bf16(a0, b0, acc[0]);
    acc[1] = wmma_bf16(a0, b1, acc[1]);
    acc[2] = wmma_bf16(a1, b0, acc[2]);
    acc[3] = wmma_bf16(a1, b1, acc[3]);
    __syncthreads();
  }

  int hi = lane >> 4, ln15 = lane & 15;
#pragma unroll
  for (int t = 0; t < 4; ++t) {
    int tm = t >> 1, tn = t & 1;
#pragma unroll
    for (int r = 0; r < 8; ++r) {
      int m = bm0 + wm * 32 + tm * 16 + r + 8 * hi;
      int n = bn0 + wn * 32 + tn * 16 + ln15;
      float vv = acc[t][r] * alpha;
      if (OUT_BF16) ((unsigned short*)Cp)[(size_t)m * ldc + n] = f2bf(vv);
      else          ((float*)Cp)[(size_t)m * ldc + n] = vv;
    }
  }
}

// ---------- fused masked attention: sim = qK^T, mask, softmax, P@V ----------
__global__ __launch_bounds__(256) void attn_kernel(const unsigned short* __restrict__ q,
                                                   const unsigned short* __restrict__ kws,
                                                   const unsigned short* __restrict__ vT,
                                                   const unsigned char* __restrict__ mmask,
                                                   const int* __restrict__ endi,
                                                   unsigned short* __restrict__ ao) {
  int bt = blockIdx.x, h = blockIdx.y, b = blockIdx.z;
  __shared__ __attribute__((aligned(16))) unsigned short sK[L_ * DH_];  // [l][dh]
  __shared__ __attribute__((aligned(16))) unsigned short sV[DH_ * L_];  // [dh][l]
  __shared__ __attribute__((aligned(16))) unsigned short sP[8 * 16 * L_];

  const unsigned short* ksrc = kws + ((size_t)b * H_ + h) * L_ * DH_;
  const unsigned short* vsrc = vT  + ((size_t)b * H_ + h) * DH_ * L_;
#pragma unroll
  for (int c = threadIdx.x; c < 512; c += 256) {
    async_ld16(sK + c * 8, ksrc + c * 8);
    async_ld16(sV + c * 8, vsrc + c * 8);
  }
  wait_async0();
  __syncthreads();

  int w = threadIdx.x >> 5, lane = threadIdx.x & 31;
  int hi = lane >> 4, ln15 = lane & 15;
  int t0 = bt * 128 + w * 16;

  // sim = q @ K^T   (K-dim = DH = 64 -> 2 WMMA K-steps; N = L = 64 -> 4 tiles)
  v8f accS[4];
#pragma unroll
  for (int t = 0; t < 4; ++t)
#pragma unroll
    for (int r = 0; r < 8; ++r) accS[t][r] = 0.0f;

  const unsigned short* qbase = q + (size_t)(b * T_ + t0) * INNER_;
#pragma unroll
  for (int ks = 0; ks < 2; ++ks) {
    v16bf aF = fragA(qbase, INNER_, 0, h * DH_ + ks * 32);
#pragma unroll
    for (int nt = 0; nt < 4; ++nt) {
      v16bf bF = fragB(sK, DH_, nt * 16, ks * 32);
      accS[nt] = wmma_bf16(aF, bF, accS[nt]);
    }
  }

  // mask: media valid AND j < end[b][t]
  int er[8];
#pragma unroll
  for (int r = 0; r < 8; ++r) er[r] = endi[b * T_ + t0 + r + 8 * hi];
#pragma unroll
  for (int nt = 0; nt < 4; ++nt) {
    int j = nt * 16 + ln15;
    bool mv = mmask[b * L_ + j] != 0;
#pragma unroll
    for (int r = 0; r < 8; ++r) {
      bool ok = mv && (j < er[r]);
      accS[nt][r] = ok ? accS[nt][r] : -3.4028235e38f;
    }
  }

  // row-wise softmax over 64 columns (values live in a 16-lane group)
#pragma unroll
  for (int r = 0; r < 8; ++r) {
    float m = accS[0][r];
#pragma unroll
    for (int nt = 1; nt < 4; ++nt) m = fmaxf(m, accS[nt][r]);
#pragma unroll
    for (int off = 8; off >= 1; off >>= 1) m = fmaxf(m, __shfl_xor(m, off, 32));
    float s = 0.f;
#pragma unroll
    for (int nt = 0; nt < 4; ++nt) {
      float e = __expf(accS[nt][r] - m);
      accS[nt][r] = e; s += e;
    }
#pragma unroll
    for (int off = 8; off >= 1; off >>= 1) s += __shfl_xor(s, off, 32);
    float is = 1.0f / s;
#pragma unroll
    for (int nt = 0; nt < 4; ++nt) accS[nt][r] *= is;
  }

  // stage P (wave-private) in A-operand friendly row-major [16][64]
  unsigned short* sPw = sP + w * 16 * L_;
#pragma unroll
  for (int nt = 0; nt < 4; ++nt)
#pragma unroll
    for (int r = 0; r < 8; ++r)
      sPw[(r + 8 * hi) * L_ + nt * 16 + ln15] = f2bf(accS[nt][r]);

  // out = P @ V   (K-dim = L = 64 -> 2 steps; N = DH = 64 -> 4 tiles)
  v8f accO[4];
#pragma unroll
  for (int t = 0; t < 4; ++t)
#pragma unroll
    for (int r = 0; r < 8; ++r) accO[t][r] = 0.0f;
#pragma unroll
  for (int ks = 0; ks < 2; ++ks) {
    v16bf aF = fragA(sPw, L_, 0, ks * 32);
#pragma unroll
    for (int nt = 0; nt < 4; ++nt) {
      v16bf bF = fragB(sV, L_, nt * 16, ks * 32);
      accO[nt] = wmma_bf16(aF, bF, accO[nt]);
    }
  }

#pragma unroll
  for (int nt = 0; nt < 4; ++nt)
#pragma unroll
    for (int r = 0; r < 8; ++r) {
      int t = t0 + r + 8 * hi;
      ao[(size_t)(b * T_ + t) * INNER_ + h * DH_ + nt * 16 + ln15] = f2bf(accO[nt][r]);
    }
}

// ---------- launch ----------
extern "C" void kernel_launch(void* const* d_in, const int* in_sizes, int n_in,
                              void* d_out, int out_size, void* d_ws, size_t ws_size,
                              hipStream_t stream) {
  (void)in_sizes; (void)n_in; (void)out_size; (void)ws_size;
  const float*         x     = (const float*)d_in[0];
  const float*         media = (const float*)d_in[1];
  const unsigned char* mmask = (const unsigned char*)d_in[2];
  const int*           mloc  = (const int*)d_in[3];
  const float*         Wq    = (const float*)d_in[4];
  const float*         Wkv   = (const float*)d_in[5];
  const float*         Wout  = (const float*)d_in[6];
  const float*         gamma = (const float*)d_in[7];
  const float*         beta  = (const float*)d_in[8];
  float* out = (float*)d_out;

  char* base = (char*)d_ws;
  size_t off = 0;
  auto alloc = [&](size_t bytes) {
    void* p = base + off;
    off += (bytes + 255) & ~(size_t)255;
    return p;
  };
  unsigned short* xn    = (unsigned short*)alloc((size_t)B_ * T_ * DIM_ * 2);    // 64 MB
  unsigned short* qb    = (unsigned short*)alloc((size_t)B_ * T_ * INNER_ * 2);  // 16 MB
  unsigned short* aob   = (unsigned short*)alloc((size_t)B_ * T_ * INNER_ * 2);  // 16 MB
  unsigned short* kb    = (unsigned short*)alloc((size_t)B_ * H_ * L_ * DH_ * 2);
  unsigned short* vtb   = (unsigned short*)alloc((size_t)B_ * H_ * DH_ * L_ * 2);
  unsigned short* wqb   = (unsigned short*)alloc((size_t)DIM_ * INNER_ * 2);
  unsigned short* woutb = (unsigned short*)alloc((size_t)INNER_ * DIM_ * 2);
  int*            endi  = (int*)alloc((size_t)B_ * T_ * 4);

  ln_kernel<<<B_ * T_, 256, 0, stream>>>(x, gamma, beta, xn);
  cvt_kernel<<<(DIM_ * INNER_ + 255) / 256, 256, 0, stream>>>(Wq, wqb, DIM_ * INNER_);
  cvt_kernel<<<(INNER_ * DIM_ + 255) / 256, 256, 0, stream>>>(Wout, woutb, INNER_ * DIM_);
  kv_kernel<<<(B_ * L_ * 1024) / 256, 256, 0, stream>>>(media, Wkv, kb, vtb);
  scan_kernel<<<B_, 256, 0, stream>>>(mloc, endi);

  // Q = LN(x) @ Wq * 1/sqrt(DH)
  gemm_kernel<true><<<dim3(INNER_ / 64, (B_ * T_) / 128), 256, 0, stream>>>(
      xn, wqb, qb, B_ * T_, INNER_, DIM_, DIM_, INNER_, INNER_, 0.125f);

  attn_kernel<<<dim3(T_ / 128, H_, B_), 256, 0, stream>>>(qb, kb, vtb, mmask, endi, aob);

  // out = ao @ Wout (fp32 out)
  gemm_kernel<false><<<dim3(DIM_ / 64, (B_ * T_) / 128), 256, 0, stream>>>(
      aob, woutb, out, B_ * T_, DIM_, INNER_, INNER_, DIM_, DIM_, 1.0f);
}
